// TransformerWorldModel_31327491457082
// MI455X (gfx1250) — compile-verified, matrix operand order checked by reading, not run on hardware
//
#include <hip/hip_runtime.h>
#include <hip/hip_bf16.h>

typedef __bf16 bf16_t;
typedef __attribute__((ext_vector_type(16))) __bf16 v16bf;
typedef __attribute__((ext_vector_type(8)))  __bf16 v8bf;
typedef __attribute__((ext_vector_type(8)))  float  v8f;

#define SEQ 2048
#define DIM 1024
#define QH 16
#define KVH 4
#define DH 64
#define NCAT 1792   // 1024 q + 256 k + 256 v + 256 vr

__device__ __forceinline__ v8f wmma_bf16(v16bf a, v16bf b, v8f c) {
  return __builtin_amdgcn_wmma_f32_16x16x32_bf16(false, a, false, b, (short)0, c, false, false);
}

// Load a 16x32 bf16 A/B fragment for this lane. `p` must point at element
// (row_for_lane, k0 + (lane>>4)*8) of a K-contiguous row: elems 0..7 come from
// p[0..7] (K = kb..kb+7), elems 8..15 from p[16..23] (K = 16+kb..16+kb+7).
__device__ __forceinline__ v16bf load_frag16(const bf16_t* __restrict__ p) {
  v8bf lo = *(const v8bf*)(p);
  v8bf hi = *(const v8bf*)(p + 16);
  v16bf f;
#pragma unroll
  for (int i = 0; i < 8; ++i) { f[i] = lo[i]; f[i + 8] = hi[i]; }
  return f;
}

// ---------------- conversion / transpose prep ----------------
__global__ void cvt_f32_to_bf16(const float* __restrict__ in, bf16_t* __restrict__ out, int n) {
  int i = blockIdx.x * blockDim.x + threadIdx.x;
  if (i < n) out[i] = (bf16_t)in[i];
}

// out[n*rowsK + k] = W[k*colsN + n]  (W is [K, N] f32, out is [N, K] bf16)
__global__ void transpose_cvt(const float* __restrict__ W, bf16_t* __restrict__ out,
                              int rowsK, int colsN) {
  int i = blockIdx.x * blockDim.x + threadIdx.x;
  if (i < rowsK * colsN) {
    int n = i / rowsK;
    int k = i - n * rowsK;
    out[i] = (bf16_t)W[(size_t)k * colsN + n];
  }
}

// ---------------- WMMA GEMM: C[M,N] f32 = A[M,K]bf16 * Bt[N,K]bf16^T ----------------
// One wave computes a 32x64 output tile: 2 A-frags + 4 B-frags -> 8 WMMAs / k-step.
// launch_bounds(32,1): single wave32 per EU target -> full VGPR budget, no spills.
__global__ void __launch_bounds__(32, 1)
gemm_bf16_wmma(const bf16_t* __restrict__ A, const bf16_t* __restrict__ Bt,
               float* __restrict__ C, int M, int N, int K) {
  int tn = blockIdx.x;          // 64-wide N tile
  int tm = blockIdx.y;          // 32-tall M tile
  int lane = threadIdx.x;       // wave32
  int m15 = lane & 15;
  int half = lane >> 4;
  int kb  = half * 8;
  int row = tm * 32;
  int col = tn * 64;
  const bf16_t* ap0 = A  + (size_t)(row + m15) * K + kb;
  const bf16_t* ap1 = ap0 + (size_t)16 * K;
  const bf16_t* bp  = Bt + (size_t)(col + m15) * K + kb;
  v8f acc[2][4] = {};
  for (int k0 = 0; k0 < K; k0 += 32) {
    // Speculative prefetch of the weight stream ~512 elements ahead; OOB
    // speculative prefetches are silently dropped by hardware.
    __builtin_prefetch(bp + (size_t)0 * 16 * K + k0 + 512, 0, 1);
    __builtin_prefetch(bp + (size_t)1 * 16 * K + k0 + 512, 0, 1);
    __builtin_prefetch(bp + (size_t)2 * 16 * K + k0 + 512, 0, 1);
    __builtin_prefetch(bp + (size_t)3 * 16 * K + k0 + 512, 0, 1);
    v16bf a0 = load_frag16(ap0 + k0);
    v16bf a1 = load_frag16(ap1 + k0);
    v16bf b0 = load_frag16(bp + (size_t)0 * 16 * K + k0);
    v16bf b1 = load_frag16(bp + (size_t)1 * 16 * K + k0);
    v16bf b2 = load_frag16(bp + (size_t)2 * 16 * K + k0);
    v16bf b3 = load_frag16(bp + (size_t)3 * 16 * K + k0);
    acc[0][0] = wmma_bf16(a0, b0, acc[0][0]);
    acc[0][1] = wmma_bf16(a0, b1, acc[0][1]);
    acc[0][2] = wmma_bf16(a0, b2, acc[0][2]);
    acc[0][3] = wmma_bf16(a0, b3, acc[0][3]);
    acc[1][0] = wmma_bf16(a1, b0, acc[1][0]);
    acc[1][1] = wmma_bf16(a1, b1, acc[1][1]);
    acc[1][2] = wmma_bf16(a1, b2, acc[1][2]);
    acc[1][3] = wmma_bf16(a1, b3, acc[1][3]);
  }
#pragma unroll
  for (int mi = 0; mi < 2; ++mi) {
    int rbase = row + mi * 16 + 8 * half;
#pragma unroll
    for (int ti = 0; ti < 4; ++ti) {
      int n = col + ti * 16 + m15;
#pragma unroll
      for (int r = 0; r < 8; ++r)
        C[(size_t)(rbase + r) * N + n] = acc[mi][ti][r];
    }
  }
}

// ---------------- gates (x@Wg) and mix (x@Wm + bm), sigmoid ----------------
__global__ void gate_mix_kernel(const float* __restrict__ x, const float* __restrict__ Wg,
                                const float* __restrict__ Wm, const float* __restrict__ bm,
                                float* __restrict__ gates, float* __restrict__ mixo) {
  int s = blockIdx.x;
  int t = threadIdx.x;
  const float* row = x + (size_t)s * DIM;
  if (t < QH) {
    float acc = 0.f;
    for (int k = 0; k < DIM; ++k) acc += row[k] * Wg[(size_t)k * QH + t];
    gates[s * QH + t] = 1.f / (1.f + __expf(-acc));
  } else if (t < QH + KVH) {
    int h = t - QH;
    float acc = bm[h];
    for (int k = 0; k < DIM; ++k) acc += row[k] * Wm[(size_t)k * KVH + h];
    mixo[s * KVH + h] = 1.f / (1.f + __expf(-acc));
  }
}

// ---------------- epilogue: RoPE / RMSNorm / value-mix ----------------
// Q -> [QH, S, DH] (pre-scaled), K -> [KVH, S, DH], V -> transposed [KVH, DH, S]
__global__ void qkv_epilogue(const float* __restrict__ Y1,        // [S, NCAT]
                             const int*   __restrict__ positions, // [S]
                             const float* __restrict__ kscale,    // [KVH, DH]
                             const float* __restrict__ mixo,      // [S, KVH]
                             bf16_t* __restrict__ Qo,             // [QH, S, DH]
                             bf16_t* __restrict__ Ko,             // [KVH, S, DH]
                             bf16_t* __restrict__ Vt) {           // [KVH, DH, S]
  int s = blockIdx.x;
  int which = blockIdx.y;     // 0..15 q-heads, 16..19 k-heads, 20..23 v-heads
  int lane = threadIdx.x;     // handles dim pair (2*lane, 2*lane+1)
  const float* row = Y1 + (size_t)s * NCAT;
  float pos = (float)positions[s];
  float inv_freq = __powf(10000.f, -(float)(2 * lane) / (float)DH);
  float fr = pos * inv_freq;
  float c = __cosf(fr), sn = __sinf(fr);
  int d0 = 2 * lane, d1 = 2 * lane + 1;

  if (which < QH) {
    int hq = which;
    float x0 = row[hq * DH + d0], x1 = row[hq * DH + d1];
    float r0 = x0 * c - x1 * sn;
    float r1 = x1 * c + x0 * sn;
    const float qscale = 0.125f;   // dh^-0.5 folded into q
    size_t base = ((size_t)hq * SEQ + s) * DH;
    Qo[base + d0] = (bf16_t)(r0 * qscale);
    Qo[base + d1] = (bf16_t)(r1 * qscale);
  } else if (which < QH + KVH) {
    int h = which - QH;
    float x0 = row[DIM + h * DH + d0], x1 = row[DIM + h * DH + d1];
    float ss = x0 * x0 + x1 * x1;
#pragma unroll
    for (int off = 1; off < 32; off <<= 1) ss += __shfl_xor(ss, off, 32);
    float inv = rsqrtf(ss / (float)DH + 1e-8f);
    x0 = x0 * inv * kscale[h * DH + d0];
    x1 = x1 * inv * kscale[h * DH + d1];
    float r0 = x0 * c - x1 * sn;
    float r1 = x1 * c + x0 * sn;
    size_t base = ((size_t)h * SEQ + s) * DH;
    Ko[base + d0] = (bf16_t)r0;
    Ko[base + d1] = (bf16_t)r1;
  } else {
    int h = which - QH - KVH;
    float mv = mixo[s * KVH + h];
    float v0 = row[DIM + 256 + h * DH + d0];
    float v1 = row[DIM + 256 + h * DH + d1];
    float w0 = row[DIM + 512 + h * DH + d0];
    float w1 = row[DIM + 512 + h * DH + d1];
    v0 = v0 + mv * (w0 - v0);
    v1 = v1 + mv * (w1 - v1);
    Vt[((size_t)h * DH + d0) * SEQ + s] = (bf16_t)v0;
    Vt[((size_t)h * DH + d1) * SEQ + s] = (bf16_t)v1;
  }
}

// ---------------- flash-style attention, one wave per (q-head, 16-query tile) ----
// Processes 32 keys per iteration: 4 WMMA for S (two 16x16 tiles), full-K 16x32 P
// fragment via LDS, 4 full-K WMMA for PV using transposed V.
__global__ void __launch_bounds__(32, 1)
attention_kernel(const bf16_t* __restrict__ Q,   // [QH, S, DH] pre-scaled
                 const bf16_t* __restrict__ Kb,  // [KVH, S, DH]
                 const bf16_t* __restrict__ Vt,  // [KVH, DH, S]
                 const float* __restrict__ gates,// [S, QH]
                 bf16_t* __restrict__ Out) {     // [S, QH*DH]
  __shared__ bf16_t pl[16 * 32];
  int qt = blockIdx.x;
  int hq = blockIdx.y;
  int hk = hq >> 2;           // 4 q-heads per kv-head
  int lane = threadIdx.x;
  int m15 = lane & 15;
  int half = lane >> 4;
  int kb = half * 8;
  int qrow = qt * 16;

  const bf16_t* Qh  = Q  + (size_t)hq * SEQ * DH;
  const bf16_t* Kh  = Kb + (size_t)hk * SEQ * DH;
  const bf16_t* Vth = Vt + (size_t)hk * DH * SEQ;

  v16bf aq0 = load_frag16(Qh + (size_t)(qrow + m15) * DH + 0  + kb);
  v16bf aq1 = load_frag16(Qh + (size_t)(qrow + m15) * DH + 32 + kb);

  v8f o[4] = {v8f{}, v8f{}, v8f{}, v8f{}};
  float m[8], l[8];
#pragma unroll
  for (int r = 0; r < 8; ++r) { m[r] = -1e30f; l[r] = 0.f; }

  int nk32 = (qt >> 1) + 1;   // 32-key blocks covering keys 0..qrow+15
  for (int j = 0; j < nk32; ++j) {
    int krow = j * 32;
    // ---- S = Q K^T for two 16-key tiles ----
    v16bf bk0a = load_frag16(Kh + (size_t)(krow +      m15) * DH + 0  + kb);
    v16bf bk0b = load_frag16(Kh + (size_t)(krow +      m15) * DH + 32 + kb);
    v16bf bk1a = load_frag16(Kh + (size_t)(krow + 16 + m15) * DH + 0  + kb);
    v16bf bk1b = load_frag16(Kh + (size_t)(krow + 16 + m15) * DH + 32 + kb);
    v8f s0 = {}, s1 = {};
    s0 = wmma_bf16(aq0, bk0a, s0);
    s0 = wmma_bf16(aq1, bk0b, s0);
    s1 = wmma_bf16(aq0, bk1a, s1);
    s1 = wmma_bf16(aq1, bk1b, s1);

    // ---- softclamp + causal mask (scale folded into Q) ----
    int key0 = krow + m15;
    int key1 = krow + 16 + m15;
    float sv0[8], sv1[8];
#pragma unroll
    for (int r = 0; r < 8; ++r) {
      int qr = qrow + r + 8 * half;
      float x0 = 50.f * tanhf(s0[r] * 0.02f);
      float x1 = 50.f * tanhf(s1[r] * 0.02f);
      sv0[r] = (key0 <= qr) ? x0 : -3.0e38f;
      sv1[r] = (key1 <= qr) ? x1 : -3.0e38f;
    }

    // ---- online softmax across 32 keys ----
    float p0[8], p1[8];
#pragma unroll
    for (int r = 0; r < 8; ++r) {
      float vmx = fmaxf(sv0[r], sv1[r]);
#pragma unroll
      for (int off = 1; off < 16; off <<= 1) vmx = fmaxf(vmx, __shfl_xor(vmx, off, 32));
      float nm = fmaxf(m[r], vmx);
      p0[r] = __expf(sv0[r] - nm);
      p1[r] = __expf(sv1[r] - nm);
      float rs = p0[r] + p1[r];
#pragma unroll
      for (int off = 1; off < 16; off <<= 1) rs += __shfl_xor(rs, off, 32);
      float alpha = __expf(m[r] - nm);
      l[r] = l[r] * alpha + rs;
      m[r] = nm;
      o[0][r] *= alpha; o[1][r] *= alpha; o[2][r] *= alpha; o[3][r] *= alpha;
    }

    // ---- P (D-layout) -> LDS [16 rows x 32 keys] -> full-K A fragment ----
#pragma unroll
    for (int r = 0; r < 8; ++r) {
      pl[(r + 8 * half) * 32 + m15]      = (bf16_t)p0[r];
      pl[(r + 8 * half) * 32 + 16 + m15] = (bf16_t)p1[r];
    }
    __syncthreads();
    v16bf ap = load_frag16(&pl[m15 * 32 + kb]);
    __syncthreads();

    // ---- PV: 4 dh-tiles, full K=32, contiguous key loads from Vt ----
#pragma unroll
    for (int t = 0; t < 4; ++t) {
      v16bf bv = load_frag16(Vth + (size_t)(t * 16 + m15) * SEQ + krow + kb);
      o[t] = wmma_bf16(ap, bv, o[t]);
    }
  }

  // ---- epilogue: 1/l, gate, store bf16 ----
#pragma unroll
  for (int r = 0; r < 8; ++r) {
    int qr = qrow + r + 8 * half;
    float g = gates[qr * QH + hq];
    float inv = g / l[r];
    size_t base = (size_t)qr * (QH * DH) + hq * DH;
#pragma unroll
    for (int t = 0; t < 4; ++t)
      Out[base + t * 16 + m15] = (bf16_t)(o[t][r] * inv);
  }
}

// ---------------- host launcher ----------------
extern "C" void kernel_launch(void* const* d_in, const int* in_sizes, int n_in,
                              void* d_out, int out_size, void* d_ws, size_t ws_size,
                              hipStream_t stream) {
  const float* x    = (const float*)d_in[0];
  const int*   pos  = (const int*)  d_in[1];
  // d_in[2] = attn_mask (causal triu) -- applied analytically
  const float* Wq   = (const float*)d_in[3];
  const float* Wk   = (const float*)d_in[4];
  const float* Wv   = (const float*)d_in[5];
  const float* Wo   = (const float*)d_in[6];
  const float* Wg   = (const float*)d_in[7];
  const float* Wm   = (const float*)d_in[8];
  const float* bm   = (const float*)d_in[9];
  const float* Wvr  = (const float*)d_in[10];
  const float* ksc  = (const float*)d_in[11];
  float* out = (float*)d_out;

  char* ws = (char*)d_ws;
  size_t off = 0;
  auto carve = [&](size_t bytes) -> char* {
    char* p = ws + off;
    off = (off + bytes + 255) & ~(size_t)255;
    return p;
  };
  bf16_t* xb    = (bf16_t*)carve((size_t)SEQ * DIM * 2);
  bf16_t* WcatT = (bf16_t*)carve((size_t)NCAT * DIM * 2);
  bf16_t* WoT   = (bf16_t*)carve((size_t)DIM * DIM * 2);
  float*  Y1    = (float*) carve((size_t)SEQ * NCAT * 4);
  bf16_t* qb    = (bf16_t*)carve((size_t)QH  * SEQ * DH * 2);
  bf16_t* kbuf  = (bf16_t*)carve((size_t)KVH * SEQ * DH * 2);
  bf16_t* vtb   = (bf16_t*)carve((size_t)KVH * DH * SEQ * 2);
  float*  mixo  = (float*) carve((size_t)SEQ * KVH * 4);
  float*  gat   = (float*) carve((size_t)SEQ * QH * 4);
  bf16_t* attn  = (bf16_t*)carve((size_t)SEQ * QH * DH * 2);

  // 1) convert x to bf16
  {
    int n = SEQ * DIM;
    cvt_f32_to_bf16<<<(n + 255) / 256, 256, 0, stream>>>(x, xb, n);
  }
  // 2) transpose+convert weights into [N, K] bf16 (concatenated q|k|v|vr, plus Wo^T)
  {
    int n;
    n = DIM * DIM;  transpose_cvt<<<(n + 255) / 256, 256, 0, stream>>>(Wq,  WcatT,              DIM, DIM);
    n = DIM * 256;  transpose_cvt<<<(n + 255) / 256, 256, 0, stream>>>(Wk,  WcatT + 1024 * DIM, DIM, 256);
    n = DIM * 256;  transpose_cvt<<<(n + 255) / 256, 256, 0, stream>>>(Wv,  WcatT + 1280 * DIM, DIM, 256);
    n = DIM * 256;  transpose_cvt<<<(n + 255) / 256, 256, 0, stream>>>(Wvr, WcatT + 1536 * DIM, DIM, 256);
    n = DIM * DIM;  transpose_cvt<<<(n + 255) / 256, 256, 0, stream>>>(Wo,  WoT,                DIM, DIM);
  }
  // 3) fused projection GEMM: Y1[S, NCAT] = xb @ Wcat   (32x64 tiles per wave)
  gemm_bf16_wmma<<<dim3(NCAT / 64, SEQ / 32), dim3(32), 0, stream>>>(xb, WcatT, Y1, SEQ, NCAT, DIM);
  // 4) gates + mix
  gate_mix_kernel<<<dim3(SEQ), dim3(32), 0, stream>>>(x, Wg, Wm, bm, gat, mixo);
  // 5) RoPE / RMSNorm / value-mix epilogue (V written transposed)
  qkv_epilogue<<<dim3(SEQ, QH + 2 * KVH), dim3(32), 0, stream>>>(Y1, pos, ksc, mixo, qb, kbuf, vtb);
  // 6) attention
  attention_kernel<<<dim3(SEQ / 16, QH), dim3(32), 0, stream>>>(qb, kbuf, vtb, gat, attn);
  // 7) output projection: out[S, DIM] = attn @ Wo
  gemm_bf16_wmma<<<dim3(DIM / 64, SEQ / 32), dim3(32), 0, stream>>>(attn, WoT, out, SEQ, DIM, DIM);

  (void)in_sizes; (void)n_in; (void)out_size; (void)ws_size;
}
